// MoEFFN_55551107006930
// MI455X (gfx1250) — matrix operand order repaired and checked by reading, not compiled, not run on hardware
//
#include <hip/hip_runtime.h>
#include <hip/hip_bf16.h>

// ---------------------------------------------------------------------------
// MoE FFN for MI455X (gfx1250): top-2 routed experts + shared expert.
// bf16 WMMA (v_wmma_f32_16x16x32_bf16), 64x128 block tiles with 4x M register
// blocking; A panels staged via GLOBAL_LOAD_ASYNC_TO_LDS_B128 (ASYNCcnt).
// ---------------------------------------------------------------------------

#define NTOK   4096      // B*T
#define HDIM   1024
#define IDIM   2048
#define I2DIM  1024
#define NEXP   8
#define TOPK   2
#define NSLOT  (NTOK * TOPK)            // 8192 routed rows
#define SLOTCAP (NSLOT + 64 * NEXP)     // 64-aligned padding per expert = 8704
#define TILES64 (SLOTCAP / 64)          // 136
#define KPANEL 256                      // LDS K-panel width (bf16 elems)

typedef __attribute__((ext_vector_type(16))) __bf16 v16bf;
typedef __attribute__((ext_vector_type(8)))  __bf16 v8bf;
typedef __attribute__((ext_vector_type(8)))  float  v8f;

__device__ __forceinline__ __bf16 f2bf(float f) {
    unsigned u = __builtin_bit_cast(unsigned, f);
    unsigned r = (u + 0x7FFFu + ((u >> 16) & 1u)) >> 16;   // round-nearest-even
    unsigned short h = (unsigned short)r;
    return __builtin_bit_cast(__bf16, h);
}

// 16-byte global -> LDS async DMA (no VGPR round-trip); tracked by ASYNCcnt.
__device__ __forceinline__ void async_copy16(unsigned lds_addr, const void* gsrc) {
    asm volatile("global_load_async_to_lds_b128 %0, %1, off"
                 :: "v"(lds_addr),
                    "v"((unsigned long long)(uintptr_t)gsrc)
                 : "memory");
}

__device__ __forceinline__ void wait_async0() {
#if __has_builtin(__builtin_amdgcn_s_wait_asynccnt)
    __builtin_amdgcn_s_wait_asynccnt(0);
#else
    asm volatile("s_wait_asynccnt 0x0" ::: "memory");
#endif
}

// Low 32 bits of a generic pointer to LDS == wave-relative LDS byte address.
__device__ __forceinline__ unsigned lds_addr_of(const void* p) {
    return (unsigned)(unsigned long long)(uintptr_t)p;
}

// --------------------------- small helper kernels ---------------------------

__global__ void init_kernel(int* __restrict__ tok_of_slot, int* __restrict__ stats) {
    int i = blockIdx.x * blockDim.x + threadIdx.x;
    if (i < SLOTCAP) tok_of_slot[i] = 0;
    if (i < 64) stats[i] = 0;
}

__global__ void cvt_bf16_kernel(const float* __restrict__ src,
                                __bf16* __restrict__ dst, size_t n) {
    size_t i = (size_t)blockIdx.x * blockDim.x + threadIdx.x;
    size_t stride = (size_t)gridDim.x * blockDim.x;
    for (; i < n; i += stride) dst[i] = f2bf(src[i]);
}

// stats layout: [0..7] counts, [8..15] cursors, [16..24] offsets, [32..39] prob_sum(f32)
__global__ __launch_bounds__(256)
void router_kernel(const float* __restrict__ X, const float* __restrict__ RW,
                   int* __restrict__ topi, float* __restrict__ topw,
                   int* __restrict__ stats) {
    __shared__ float part[NEXP][32];
    __shared__ float logit[NEXP];
    int n = blockIdx.x;
    int e = threadIdx.x & 7;
    int c = threadIdx.x >> 3;                 // 0..31
    const float* x  = X  + (size_t)n * HDIM + c * 32;
    const float* rw = RW + (size_t)e * HDIM + c * 32;
    float p = 0.f;
    #pragma unroll 8
    for (int j = 0; j < 32; ++j) p += x[j] * rw[j];
    part[e][c] = p;
    __syncthreads();
    if (threadIdx.x < NEXP) {
        float s = 0.f;
        for (int j = 0; j < 32; ++j) s += part[threadIdx.x][j];
        logit[threadIdx.x] = s;
    }
    __syncthreads();
    if (threadIdx.x == 0) {
        float mx = logit[0];
        for (int j = 1; j < NEXP; ++j) mx = fmaxf(mx, logit[j]);
        float pe[NEXP]; float den = 0.f;
        for (int j = 0; j < NEXP; ++j) { pe[j] = __expf(logit[j] - mx); den += pe[j]; }
        float inv = 1.f / den;
        int i0 = 0;
        for (int j = 0; j < NEXP; ++j) { pe[j] *= inv; if (pe[j] > pe[i0]) i0 = j; }
        int i1 = (i0 == 0) ? 1 : 0;
        for (int j = 0; j < NEXP; ++j) if (j != i0 && pe[j] > pe[i1]) i1 = j;
        float w0 = pe[i0], w1 = pe[i1], sw = 1.f / (w0 + w1);
        topi[n * 2 + 0] = i0;  topi[n * 2 + 1] = i1;
        topw[n * 2 + 0] = w0 * sw;  topw[n * 2 + 1] = w1 * sw;
        atomicAdd(&stats[i0], 1);
        atomicAdd(&stats[i1], 1);
        float* prob_sum = (float*)(stats + 32);
        for (int j = 0; j < NEXP; ++j) atomicAdd(&prob_sum[j], pe[j]);
    }
}

__global__ void offsets_kernel(int* __restrict__ stats) {
    // single thread: 64-aligned exclusive scan of counts
    int* counts = stats;
    int* cursor = stats + 8;
    int* off    = stats + 16;
    int acc = 0;
    for (int e = 0; e < NEXP; ++e) {
        off[e] = acc;
        cursor[e] = acc;
        acc += (counts[e] + 63) & ~63;
    }
    off[NEXP] = acc;
}

__global__ void scatter_kernel(const int* __restrict__ topi,
                               int* __restrict__ stats,
                               int* __restrict__ tok_of_slot,
                               int* __restrict__ slot_of) {
    int n = blockIdx.x * blockDim.x + threadIdx.x;
    if (n >= NTOK) return;
    int* cursor = stats + 8;
    for (int k = 0; k < TOPK; ++k) {
        int e = topi[n * 2 + k];
        int slot = atomicAdd(&cursor[e], 1);
        tok_of_slot[slot] = n;
        slot_of[n * 2 + k] = slot;
    }
}

// ------------------------------ GEMM kernels -------------------------------
// Block: 256 threads = 8 waves. Each block computes a 64(row) x 128(col) tile.
// Each wave owns one 16-col slice and 4 M-subtiles: one B fragment per K-step
// is reused by 4 WMMAs (8 for fused gate+up). A panels (64 x 256 bf16, 33 KB)
// are staged with async global->LDS DMA, drained via s_wait_asynccnt.

__global__ __launch_bounds__(256)
void moe_gateup_kernel(const __bf16* __restrict__ X,      // [NTOK, HDIM] bf16
                       const int* __restrict__ tokmap,    // slot -> token, or null
                       const int* __restrict__ offs,      // [E+1] or null (shared)
                       const __bf16* __restrict__ Wg,     // [(E,)Id,H]
                       const __bf16* __restrict__ Wu,     // [(E,)Id,H]
                       __bf16* __restrict__ Hid,          // [rows, Id]
                       int Id) {
    __shared__ __align__(16) __bf16 As[64][KPANEL + 8];
    int row_base = blockIdx.x * 64;
    int e = 0;
    if (offs) {
        if (row_base >= offs[NEXP]) return;
        while (row_base >= offs[e + 1]) ++e;
    }
    int ibase = blockIdx.y * 128 + (threadIdx.x >> 5) * 16;

    int lane = threadIdx.x & 31;
    int col  = lane & 15;
    int hi   = lane >> 4;
    const __bf16* wg = Wg + (size_t)e * Id * HDIM + (size_t)(ibase + col) * HDIM;
    const __bf16* wu = Wu + (size_t)e * Id * HDIM + (size_t)(ibase + col) * HDIM;

    int r = threadIdx.x >> 2;                 // 0..63 (staging row)
    int cseg = (threadIdx.x & 3) * 64;        // 64 elems / thread
    int tok = tokmap ? tokmap[row_base + r] : row_base + r;
    const __bf16* arow = X + (size_t)tok * HDIM + cseg;
    unsigned lbase = lds_addr_of(&As[r][cseg]);

    v8f accg[4] = {}; v8f accu[4] = {};
    for (int kp = 0; kp < HDIM; kp += KPANEL) {
        __syncthreads();                       // prev panel fully consumed
        #pragma unroll
        for (int j = 0; j < 64; j += 8)        // 8 x async 16B: global -> LDS
            async_copy16(lbase + j * 2, arow + kp + j);
        wait_async0();
        __syncthreads();
        for (int kb = 0; kb < KPANEL; kb += 32) {
            v16bf bg = *(const v16bf*)&wg[kp + kb + hi * 16];
            v16bf bu = *(const v16bf*)&wu[kp + kb + hi * 16];
            #pragma unroll
            for (int mt = 0; mt < 4; ++mt) {
                union { v16bf v; v8bf h[2]; } ua;
                ua.h[0] = *(const v8bf*)&As[mt * 16 + col][kb + hi * 16];
                ua.h[1] = *(const v8bf*)&As[mt * 16 + col][kb + hi * 16 + 8];
                v16bf a = ua.v;
                accg[mt] = __builtin_amdgcn_wmma_f32_16x16x32_bf16(
                    false, a, false, bg, (short)0, accg[mt], false, false);
                accu[mt] = __builtin_amdgcn_wmma_f32_16x16x32_bf16(
                    false, a, false, bu, (short)0, accu[mt], false, false);
            }
        }
    }
    // silu(g) * u -> Hid (bf16)
    #pragma unroll
    for (int mt = 0; mt < 4; ++mt) {
        #pragma unroll
        for (int v = 0; v < 8; ++v) {
            int m = mt * 16 + v + 8 * hi;
            float g = accg[mt][v], u = accu[mt][v];
            float s = g / (1.0f + __expf(-g));
            Hid[(size_t)(row_base + m) * Id + ibase + col] = f2bf(s * u);
        }
    }
}

__global__ __launch_bounds__(256)
void moe_down_kernel(const __bf16* __restrict__ Hid,     // [rows, Kd] bf16
                     const int* __restrict__ offs,       // [E+1] or null (shared)
                     const __bf16* __restrict__ Wd,      // [(E,)H,Kd]
                     float* __restrict__ Out,            // [rows, H]
                     int Kd, int accumulate) {
    __shared__ __align__(16) __bf16 As[64][KPANEL + 8];
    int row_base = blockIdx.x * 64;
    int e = 0;
    if (offs) {
        if (row_base >= offs[NEXP]) return;
        while (row_base >= offs[e + 1]) ++e;
    }
    int hbase = blockIdx.y * 128 + (threadIdx.x >> 5) * 16;
    int lane = threadIdx.x & 31;
    int col  = lane & 15;
    int hi   = lane >> 4;
    const __bf16* wd = Wd + (size_t)e * HDIM * Kd + (size_t)(hbase + col) * Kd;

    int r = threadIdx.x >> 2;
    int cseg = (threadIdx.x & 3) * 64;
    const __bf16* arow = Hid + (size_t)(row_base + r) * Kd + cseg;
    unsigned lbase = lds_addr_of(&As[r][cseg]);

    v8f acc[4] = {};
    for (int kp = 0; kp < Kd; kp += KPANEL) {
        __syncthreads();
        #pragma unroll
        for (int j = 0; j < 64; j += 8)
            async_copy16(lbase + j * 2, arow + kp + j);
        wait_async0();
        __syncthreads();
        for (int kb = 0; kb < KPANEL; kb += 32) {
            v16bf b = *(const v16bf*)&wd[kp + kb + hi * 16];
            #pragma unroll
            for (int mt = 0; mt < 4; ++mt) {
                union { v16bf v; v8bf h[2]; } ua;
                ua.h[0] = *(const v8bf*)&As[mt * 16 + col][kb + hi * 16];
                ua.h[1] = *(const v8bf*)&As[mt * 16 + col][kb + hi * 16 + 8];
                v16bf a = ua.v;
                acc[mt] = __builtin_amdgcn_wmma_f32_16x16x32_bf16(
                    false, a, false, b, (short)0, acc[mt], false, false);
            }
        }
    }
    #pragma unroll
    for (int mt = 0; mt < 4; ++mt) {
        #pragma unroll
        for (int v = 0; v < 8; ++v) {
            int m = mt * 16 + v + 8 * hi;
            size_t idx = (size_t)(row_base + m) * HDIM + hbase + col;
            Out[idx] = accumulate ? (Out[idx] + acc[mt][v]) : acc[mt][v];
        }
    }
}

__global__ __launch_bounds__(256)
void combine_kernel(const float* __restrict__ Y, const int* __restrict__ slot_of,
                    const float* __restrict__ topw, float* __restrict__ Out) {
    int n = blockIdx.x;
    int h = blockIdx.y * 256 + threadIdx.x;
    int s0 = slot_of[n * 2 + 0], s1 = slot_of[n * 2 + 1];
    float w0 = topw[n * 2 + 0], w1 = topw[n * 2 + 1];
    Out[(size_t)n * HDIM + h] =
        w0 * Y[(size_t)s0 * HDIM + h] + w1 * Y[(size_t)s1 * HDIM + h];
}

__global__ void aux_kernel(const int* __restrict__ stats, float* __restrict__ aux_out) {
    const float* prob_sum = (const float*)(stats + 32);
    float aux = 0.f;
    for (int e = 0; e < NEXP; ++e)
        aux += ((float)stats[e] / (float)NSLOT) * (prob_sum[e] / (float)NTOK);
    aux_out[0] = 0.01f * (float)NEXP * aux;
}

// ------------------------------- launcher ----------------------------------

static inline void* carve(char*& p, size_t bytes) {
    void* r = p;
    p += (bytes + 255) & ~(size_t)255;
    return r;
}

extern "C" void kernel_launch(void* const* d_in, const int* in_sizes, int n_in,
                              void* d_out, int out_size, void* d_ws, size_t ws_size,
                              hipStream_t stream) {
    const float* x   = (const float*)d_in[0];   // [N,H]
    const float* rw  = (const float*)d_in[1];   // [E,H]
    const float* gw  = (const float*)d_in[2];   // [E,I,H]
    const float* uw  = (const float*)d_in[3];   // [E,I,H]
    const float* dw  = (const float*)d_in[4];   // [E,H,I]
    const float* sgw = (const float*)d_in[5];   // [I2,H]
    const float* suw = (const float*)d_in[6];   // [I2,H]
    const float* sdw = (const float*)d_in[7];   // [H,I2]
    float* out = (float*)d_out;                 // [N*H] + aux scalar

    char* p = (char*)d_ws;
    __bf16* xb   = (__bf16*)carve(p, (size_t)NTOK * HDIM * 2);
    __bf16* gb   = (__bf16*)carve(p, (size_t)NEXP * IDIM * HDIM * 2);
    __bf16* ub   = (__bf16*)carve(p, (size_t)NEXP * IDIM * HDIM * 2);
    __bf16* db   = (__bf16*)carve(p, (size_t)NEXP * HDIM * IDIM * 2);
    __bf16* sgb  = (__bf16*)carve(p, (size_t)I2DIM * HDIM * 2);
    __bf16* sub  = (__bf16*)carve(p, (size_t)I2DIM * HDIM * 2);
    __bf16* sdb  = (__bf16*)carve(p, (size_t)HDIM * I2DIM * 2);
    __bf16* hid  = (__bf16*)carve(p, (size_t)SLOTCAP * IDIM * 2);
    __bf16* hids = (__bf16*)carve(p, (size_t)NTOK * I2DIM * 2);
    float*  y    = (float*) carve(p, (size_t)SLOTCAP * HDIM * 4);
    int*    topi = (int*)   carve(p, (size_t)NTOK * 2 * 4);
    float*  topw = (float*) carve(p, (size_t)NTOK * 2 * 4);
    int*  slot_of = (int*)  carve(p, (size_t)NTOK * 2 * 4);
    int*  tok_of  = (int*)  carve(p, (size_t)SLOTCAP * 4);
    int*  stats   = (int*)  carve(p, 64 * 4);
    int*  offs    = stats + 16;

    // 1) init + routing
    init_kernel<<<(SLOTCAP + 255) / 256, 256, 0, stream>>>(tok_of, stats);
    router_kernel<<<NTOK, 256, 0, stream>>>(x, rw, topi, topw, stats);
    offsets_kernel<<<1, 1, 0, stream>>>(stats);
    scatter_kernel<<<(NTOK + 255) / 256, 256, 0, stream>>>(topi, stats, tok_of, slot_of);

    // 2) f32 -> bf16 conversion (weights become L2-resident for GEMM phase)
    cvt_bf16_kernel<<<1024, 256, 0, stream>>>(x,   xb,  (size_t)NTOK * HDIM);
    cvt_bf16_kernel<<<4096, 256, 0, stream>>>(gw,  gb,  (size_t)NEXP * IDIM * HDIM);
    cvt_bf16_kernel<<<4096, 256, 0, stream>>>(uw,  ub,  (size_t)NEXP * IDIM * HDIM);
    cvt_bf16_kernel<<<4096, 256, 0, stream>>>(dw,  db,  (size_t)NEXP * HDIM * IDIM);
    cvt_bf16_kernel<<<512,  256, 0, stream>>>(sgw, sgb, (size_t)I2DIM * HDIM);
    cvt_bf16_kernel<<<512,  256, 0, stream>>>(suw, sub, (size_t)I2DIM * HDIM);
    cvt_bf16_kernel<<<512,  256, 0, stream>>>(sdw, sdb, (size_t)HDIM * I2DIM);

    // 3) routed experts: gate/up + SiLU, then down into per-slot rows
    dim3 g1(TILES64, IDIM / 128);
    moe_gateup_kernel<<<g1, 256, 0, stream>>>(xb, tok_of, offs, gb, ub, hid, IDIM);
    dim3 g2(TILES64, HDIM / 128);
    moe_down_kernel<<<g2, 256, 0, stream>>>(hid, offs, db, y, IDIM, 0);

    // 4) shared expert gate/up (identity token map)
    dim3 g1s(NTOK / 64, I2DIM / 128);
    moe_gateup_kernel<<<g1s, 256, 0, stream>>>(xb, nullptr, nullptr, sgb, sub, hids, I2DIM);

    // 5) weighted top-2 combine (deterministic, no atomics on out)
    combine_kernel<<<dim3(NTOK, HDIM / 256), 256, 0, stream>>>(y, slot_of, topw, out);

    // 6) shared expert down, accumulated into out
    dim3 g2s(NTOK / 64, HDIM / 128);
    moe_down_kernel<<<g2s, 256, 0, stream>>>(hids, nullptr, sdb, out, I2DIM, 1);

    // 7) aux loss scalar
    aux_kernel<<<1, 1, 0, stream>>>(stats, out + (size_t)NTOK * HDIM);
}
